// LSTM2Layer_3066606649426
// MI455X (gfx1250) — compile-verified, matrix operand order checked by reading, not run on hardware
//
#include <hip/hip_runtime.h>
#include <hip/hip_bf16.h>
#include <math.h>

typedef __attribute__((ext_vector_type(16))) _Float16 v16h;
typedef __attribute__((ext_vector_type(8)))  float    v8f;

#define B_TOT   512
#define T_TOT   1024
#define F_DIM   128
#define U_DIM   256
#define N_DIM   (4*U_DIM)        // 1024 gate columns
#define BT      16               // batch rows per workgroup
#define NB_TILES (B_TOT/BT)      // 32 workgroups
#define KT_X    (F_DIM/32)       // 4 k-tiles from input projection
#define KT_H    (U_DIM/32)       // 8 k-tiles from recurrent projection
#define KT_ALL  (KT_X+KT_H)      // 12
#define NT      (N_DIM/16)       // 64 n-tiles
#define FRAG_HALVES 512          // one 16x32 (A) or 32x16 (B) f16 fragment

// ---- fragment index helpers (ISA 7.12.2, 16-bit 16x32 A layout) ----
// element (m in 0..15, kk in 0..31) lives at lane = m + ((kk&15)>=8 ? 16:0),
// half index j = (kk>=16 ? 8:0) + (kk&7), within a 512-half fragment where
// lane L owns halves [L*16 .. L*16+15].

__device__ __forceinline__ float fast_sigmoid(float x) {
    return __builtin_amdgcn_rcpf(1.0f + __expf(-x));
}
__device__ __forceinline__ float fast_tanh(float x) {
    // tanh(x) = 1 - 2/(exp(2x)+1); saturates correctly at +/-inf
    return 1.0f - 2.0f * __builtin_amdgcn_rcpf(__expf(2.0f * x) + 1.0f);
}

// -------- pack weights: [kernel(128x1024); rkernel(256x1024)] -> f16 B-fragments
__global__ void lstm_pack_w(const float* __restrict__ wk,
                            const float* __restrict__ wr,
                            _Float16* __restrict__ wpk) {
    int gid  = blockIdx.x * blockDim.x + threadIdx.x;   // one per (fragment, lane)
    int lane = gid & 31;
    int frag = gid >> 5;                                // kt*NT + nt
    if (frag >= KT_ALL * NT) return;
    int kt = frag / NT;
    int nt = frag % NT;
    int n  = nt * 16 + (lane & 15);
    _Float16* dst = wpk + (size_t)frag * FRAG_HALVES + lane * 16;
#pragma unroll
    for (int j = 0; j < 16; ++j) {
        int kk  = ((j >= 8) ? 16 : 0) + ((lane >= 16) ? 8 : 0) + (j & 7);
        int row = kt * 32 + kk;
        float w = (row < F_DIM) ? wk[(size_t)row * N_DIM + n]
                                : wr[(size_t)(row - F_DIM) * N_DIM + n];
        dst[j] = (_Float16)w;
    }
}

// -------- pack x: [B,T,F] f32 -> A-fragment-major f16: [btile][t][kt][512]
__global__ void lstm_pack_x(const float* __restrict__ x,
                            _Float16* __restrict__ xpk) {
    long gid  = (long)blockIdx.x * blockDim.x + threadIdx.x;
    int  lane = (int)(gid & 31);
    long frag = gid >> 5;                               // (btile*T + t)*KT_X + kt
    if (frag >= (long)NB_TILES * T_TOT * KT_X) return;
    int  kt    = (int)(frag % KT_X);
    long bt_t  = frag / KT_X;
    int  t     = (int)(bt_t % T_TOT);
    int  btile = (int)(bt_t / T_TOT);
    int  b     = btile * BT + (lane & 15);
    const float* src = x + ((size_t)b * T_TOT + t) * F_DIM;
    _Float16* dst = xpk + frag * FRAG_HALVES + lane * 16;
#pragma unroll
    for (int j = 0; j < 16; ++j) {
        int kk = ((j >= 8) ? 16 : 0) + ((lane >= 16) ? 8 : 0) + (j & 7);
        dst[j] = (_Float16)src[kt * 32 + kk];
    }
}

// -------- persistent recurrent kernel: one WG owns 16 batch rows for all T
__global__ __launch_bounds__(256, 1)
void lstm_scan(const _Float16* __restrict__ xpk,
               const _Float16* __restrict__ wpk,
               const float* __restrict__ bias,
               const float* __restrict__ h0,
               const float* __restrict__ c0,
               float* __restrict__ out) {
    __shared__ float    z_s[BT * N_DIM];              // 64 KB gate pre-activations
    __shared__ float    c_s[BT * U_DIM];              // 16 KB cell state
    __shared__ _Float16 hfrag[KT_H * FRAG_HALVES];    //  8 KB h as A-fragments

    const int tid   = threadIdx.x;
    const int lane  = tid & 31;
    const int wave  = tid >> 5;                       // 8 waves
    const int btile = blockIdx.x;
    const int b0    = btile * BT;

    // ---- init: h0 -> f16 fragments in LDS, c0 -> LDS (tid == u, loop over b)
    for (int b = 0; b < BT; ++b) {
        const int u = tid;                            // blockDim == U_DIM
        float hv = h0[(size_t)(b0 + b) * U_DIM + u];
        float cv = c0[(size_t)(b0 + b) * U_DIM + u];
        c_s[b * U_DIM + u] = cv;
        int kt = u >> 5, kk = u & 31;
        int l2 = b + (((kk & 15) >= 8) ? 16 : 0);
        int j2 = ((kk >= 16) ? 8 : 0) + (kk & 7);
        hfrag[kt * FRAG_HALVES + l2 * 16 + j2] = (_Float16)hv;
    }

    // per-(wave,ntile) bias broadcast: C-fragment value depends only on N=lane%16
    float biasv[8];
#pragma unroll
    for (int j = 0; j < 8; ++j)
        biasv[j] = bias[(wave * 8 + j) * 16 + (lane & 15)];

    __syncthreads();

    // Opaque uniform ZERO offset, refreshed per iteration in an SGPR: weight
    // load addresses are no longer loop-invariant (so no LICM -> no spills),
    // but pointer provenance stays addrspace(1) so loads keep the
    // global_load form (no flat_load / DScnt coupling).
    int zoff = 0;

    for (int t = 0; t < T_TOT; ++t) {
        asm volatile("" : "+s"(zoff));
        const _Float16* wp = wpk + zoff;

        // ---------------- GEMM phase: z = bias + x_t*Wk + h*Wr ----------------
        v8f acc[8];
#pragma unroll
        for (int j = 0; j < 8; ++j)
#pragma unroll
            for (int r = 0; r < 8; ++r) acc[j][r] = biasv[j];

        const _Float16* xf =
            xpk + ((size_t)(btile * T_TOT + t) * KT_X) * FRAG_HALVES;
        if (t + 1 < T_TOT)  // gfx1250 global_prefetch_b8 for next step's x tile
            __builtin_prefetch(xf + (size_t)KT_X * FRAG_HALVES + lane * 16, 0, 1);

#pragma unroll
        for (int kt = 0; kt < KT_ALL; ++kt) {
            v16h a;
            if (kt < KT_X)
                a = *(const v16h*)(xf + (size_t)kt * FRAG_HALVES + lane * 16);
            else
                a = *(const v16h*)(hfrag + (kt - KT_X) * FRAG_HALVES + lane * 16);

            const _Float16* wrow =
                wp + ((size_t)(kt * NT + wave * 8)) * FRAG_HALVES + lane * 16;
            // software pipeline, depth 2: B fragment j+2 in flight while WMMA j
            v16h bq0 = *(const v16h*)wrow;
            v16h bq1 = *(const v16h*)(wrow + FRAG_HALVES);
#pragma unroll
            for (int j = 0; j < 6; ++j) {
                v16h bq2 = *(const v16h*)(wrow + (size_t)(j + 2) * FRAG_HALVES);
                acc[j] = __builtin_amdgcn_wmma_f32_16x16x32_f16(
                    false, a, false, bq0, (short)0, acc[j], false, false);
                bq0 = bq1;
                bq1 = bq2;
            }
            acc[6] = __builtin_amdgcn_wmma_f32_16x16x32_f16(
                false, a, false, bq0, (short)0, acc[6], false, false);
            acc[7] = __builtin_amdgcn_wmma_f32_16x16x32_f16(
                false, a, false, bq1, (short)0, acc[7], false, false);
        }

        // scatter C fragments to z_s: m = r + (lane>=16 ? 8 : 0), n = n0 + lane%16
        {
            const int mbase = (lane >= 16) ? 8 : 0;
#pragma unroll
            for (int j = 0; j < 8; ++j) {
                int n = (wave * 8 + j) * 16 + (lane & 15);
#pragma unroll
                for (int r = 0; r < 8; ++r)
                    z_s[(mbase + r) * N_DIM + n] = acc[j][r];
            }
        }
        __syncthreads();

        // ---------------- elementwise gate phase ----------------
        const bool last = (t == T_TOT - 1);
        for (int b = 0; b < BT; ++b) {
            const int u = tid;
            float zi = z_s[b * N_DIM + u];
            float zf = z_s[b * N_DIM + U_DIM + u];
            float zo = z_s[b * N_DIM + 2 * U_DIM + u];
            float zg = z_s[b * N_DIM + 3 * U_DIM + u];
            float ig = fast_sigmoid(zi);
            float fg = fast_sigmoid(zf);
            float og = fast_sigmoid(zo);
            float gg = fast_tanh(zg);
            float cv = fg * c_s[b * U_DIM + u] + ig * gg;
            float hv = og * fast_tanh(cv);
            c_s[b * U_DIM + u] = cv;
            int kt = u >> 5, kk = u & 31;
            int l2 = b + (((kk & 15) >= 8) ? 16 : 0);
            int j2 = ((kk >= 16) ? 8 : 0) + (kk & 7);
            hfrag[kt * FRAG_HALVES + l2 * 16 + j2] = (_Float16)hv;
            if (last) out[(size_t)(b0 + b) * U_DIM + u] = hv;
        }
        __syncthreads();
    }
}

extern "C" void kernel_launch(void* const* d_in, const int* in_sizes, int n_in,
                              void* d_out, int out_size, void* d_ws, size_t ws_size,
                              hipStream_t stream) {
    const float* x    = (const float*)d_in[0];
    const float* wk   = (const float*)d_in[1];
    const float* wr   = (const float*)d_in[2];
    const float* bias = (const float*)d_in[3];
    const float* h0   = (const float*)d_in[4];
    const float* c0   = (const float*)d_in[5];
    float* out = (float*)d_out;

    // workspace: packed weights (768 KB) then packed x (128 MB), both 32B aligned
    _Float16* wpk = (_Float16*)d_ws;
    _Float16* xpk = wpk + (size_t)KT_ALL * NT * FRAG_HALVES;

    {
        int total = KT_ALL * NT * 32;
        lstm_pack_w<<<(total + 255) / 256, 256, 0, stream>>>(wk, wr, wpk);
    }
    {
        long total = (long)NB_TILES * T_TOT * KT_X * 32;
        lstm_pack_x<<<(int)((total + 255) / 256), 256, 0, stream>>>(x, xpk);
    }
    lstm_scan<<<NB_TILES, 256, 0, stream>>>(xpk, wpk, bias, h0, c0, out);
}
